// DiffusionTransformerLayer_56736517980637
// MI455X (gfx1250) — matrix operand order repaired and checked by reading, not compiled, not run on hardware
//
#include <hip/hip_runtime.h>
#include <hip/hip_bf16.h>
#include <math.h>

typedef __attribute__((ext_vector_type(16))) _Float16 v16h;
typedef __attribute__((ext_vector_type(8)))  _Float16 v8h;
typedef __attribute__((ext_vector_type(4)))  _Float16 v4h;
typedef __attribute__((ext_vector_type(8)))  float    v8f;

#define NSEQ   512
#define DIMX   768
#define NH     16
#define HD     48
#define ZD     128
#define HIDDEN 1536

__device__ __forceinline__ float sigmoidf_(float x) { return 1.0f / (1.0f + __expf(-x)); }

// ---------------------------------------------------------------------------
// Row LayerNorm: one 256-thread block per row. w/b optional (nullptr).
// ---------------------------------------------------------------------------
__global__ __launch_bounds__(256)
void ln_rows(const float* __restrict__ x, const float* __restrict__ w,
             const float* __restrict__ b, float* __restrict__ y, int cols)
{
    __shared__ float red[256];
    const int row = blockIdx.x;
    const float* xr = x + (long long)row * cols;
    float s = 0.f, s2 = 0.f;
    for (int c = threadIdx.x; c < cols; c += 256) { float v = xr[c]; s += v; s2 += v * v; }
    red[threadIdx.x] = s; __syncthreads();
    for (int o = 128; o > 0; o >>= 1) { if (threadIdx.x < o) red[threadIdx.x] += red[threadIdx.x + o]; __syncthreads(); }
    const float mean = red[0] / (float)cols;
    __syncthreads();
    red[threadIdx.x] = s2; __syncthreads();
    for (int o = 128; o > 0; o >>= 1) { if (threadIdx.x < o) red[threadIdx.x] += red[threadIdx.x + o]; __syncthreads(); }
    const float var = red[0] / (float)cols - mean * mean;
    const float r = rsqrtf(var + 1e-5f);
    float* yr = y + (long long)row * cols;
    for (int c = threadIdx.x; c < cols; c += 256) {
        float v = (xr[c] - mean) * r;
        if (w) v *= w[c];
        if (b) v += b[c];
        yr[c] = v;
    }
}

// ---------------------------------------------------------------------------
// WMMA GEMM: C[m,n] = alpha * sum_k A[m,k] * (TRANS_B ? B[n,k] : B[k,n])
//            (+ bias[n]) (+ addm[m,n])        fp32 in/out, f16 WMMA compute.
// Block: 256 threads = 8 waves; block tile 32(M) x 64(N); K chunk 32.
// grid.z = batch (per-head), with element strides sA/sB/sC/sAdd.
// Staging uses uniform fast path (float4 loads + vector LDS stores) when the
// tile is fully in-bounds; branch-free clamped loads otherwise.
// ---------------------------------------------------------------------------
template <bool TRANS_B>
__global__ __launch_bounds__(256)
void wmma_gemm(const float* __restrict__ A, const float* __restrict__ B,
               float* __restrict__ C, const float* __restrict__ bias,
               const float* __restrict__ addm,
               int M, int N, int K, int lda, int ldb, int ldc, int ldadd,
               long long sA, long long sB, long long sC, long long sAdd,
               float alpha)
{
    __shared__ _Float16 As[32][40];   // row stride 80B: 16B aligned
    __shared__ _Float16 Bs[64][40];   // stored transposed: Bs[n][k]

    const int tid  = threadIdx.x;
    const int lane = tid & 31;
    const int wid  = tid >> 5;   // 0..7
    const int wm   = wid >> 2;   // 0..1  (16-row tile)
    const int wn   = wid & 3;    // 0..3  (16-col tile)
    const int bz   = blockIdx.z;

    const float* Ab = A + (long long)bz * sA;
    const float* Bb = B + (long long)bz * sB;
    float*       Cb = C + (long long)bz * sC;

    const int row0 = blockIdx.y * 32;
    const int col0 = blockIdx.x * 64;

    // staging coordinates
    const int ar  = tid >> 3;           // A: 0..31 rows
    const int akq = (tid & 7) << 2;     // A: k offset 0..28
    const int bn  = TRANS_B ? (tid >> 2)      : (tid & 63);       // 0..63
    const int bkq = TRANS_B ? ((tid & 3) << 3) : ((tid >> 6) << 3); // 0,8,16,24

    const bool mFull = (row0 + 32 <= M);
    const bool nFull = (col0 + 64 <= N);

    v8f acc = {};

    for (int k0 = 0; k0 < K; k0 += 32) {
        const bool kFull = (k0 + 32 <= K);

        // ---- stage A tile (32 rows x 32 k) : 4 halves / thread ----
        if (mFull && kFull) {
            const float4 f = *(const float4*)(Ab + (long long)(row0 + ar) * lda + k0 + akq);
            v4h h;
            h[0] = (_Float16)f.x; h[1] = (_Float16)f.y;
            h[2] = (_Float16)f.z; h[3] = (_Float16)f.w;
            *(v4h*)&As[ar][akq] = h;
            if (k0 + 32 < K)
                __builtin_prefetch(Ab + (long long)(row0 + ar) * lda + k0 + 32 + akq, 0, 1);
        } else {
            const int gr = row0 + ar;
#pragma unroll
            for (int i = 0; i < 4; ++i) {
                const int gk = k0 + akq + i;
                const bool ok = (gr < M) && (gk < K);
                const long long idx = ok ? ((long long)gr * lda + gk) : 0;
                const float v = Ab[idx];
                As[ar][akq + i] = (_Float16)(ok ? v : 0.f);
            }
        }

        // ---- stage B tile transposed (64 n x 32 k) : 8 halves / thread ----
        if (TRANS_B) {
            if (nFull && kFull) {
                const float* src = Bb + (long long)(col0 + bn) * ldb + k0 + bkq;
                const float4 f0 = *(const float4*)(src);
                const float4 f1 = *(const float4*)(src + 4);
                v8h h;
                h[0] = (_Float16)f0.x; h[1] = (_Float16)f0.y;
                h[2] = (_Float16)f0.z; h[3] = (_Float16)f0.w;
                h[4] = (_Float16)f1.x; h[5] = (_Float16)f1.y;
                h[6] = (_Float16)f1.z; h[7] = (_Float16)f1.w;
                *(v8h*)&Bs[bn][bkq] = h;
                if (k0 + 32 < K)
                    __builtin_prefetch(src + 32, 0, 1);
            } else {
                const int gn = col0 + bn;
#pragma unroll
                for (int i = 0; i < 8; ++i) {
                    const int gk = k0 + bkq + i;
                    const bool ok = (gn < N) && (gk < K);
                    const long long idx = ok ? ((long long)gn * ldb + gk) : 0;
                    const float v = Bb[idx];
                    Bs[bn][bkq + i] = (_Float16)(ok ? v : 0.f);
                }
            }
        } else {
            if (nFull && kFull) {
                const float* src = Bb + (long long)(k0 + bkq) * ldb + col0 + bn;
                v8h h;
#pragma unroll
                for (int i = 0; i < 8; ++i) h[i] = (_Float16)src[(long long)i * ldb];
                *(v8h*)&Bs[bn][bkq] = h;
                if (k0 + 32 < K)
                    __builtin_prefetch(src + (long long)32 * ldb, 0, 1);
            } else {
                const int gn = col0 + bn;
#pragma unroll
                for (int i = 0; i < 8; ++i) {
                    const int gk = k0 + bkq + i;
                    const bool ok = (gn < N) && (gk < K);
                    const long long idx = ok ? ((long long)gk * ldb + gn) : 0;
                    const float v = Bb[idx];
                    Bs[bn][bkq + i] = (_Float16)(ok ? v : 0.f);
                }
            }
        }
        __syncthreads();

        // ---- build fragments per ISA lane layout (wave32) ----
        // A (16-bit 16x32): lane m=lane%16; e<8 -> k=e+8*hi ; e>=8 -> k=e+8+8*hi
        // B (16-bit 32x16, transposed in LDS): n=lane%16; e<8 -> k=e+8*hi ; e>=8 -> k=e+8+8*hi
        const int hi = lane >> 4;
        const int lm = lane & 15;
        v16h af, bf;
        {
            const _Float16* ap = &As[wm * 16 + lm][hi * 8];
#pragma unroll
            for (int i = 0; i < 8; ++i) { af[i] = ap[i]; af[i + 8] = ap[i + 16]; }
            const _Float16* bp = &Bs[wn * 16 + lm][hi * 8];
#pragma unroll
            for (int i = 0; i < 8; ++i) { bf[i] = bp[i]; bf[i + 8] = bp[i + 16]; }
        }
        acc = __builtin_amdgcn_wmma_f32_16x16x32_f16(false, af, false, bf,
                                                     (short)0, acc, false, false);
        __syncthreads();
    }

    // ---- epilogue: C/D layout -> VGPR r holds M = r + 8*hi, N = lane%16 ----
    const int hi   = lane >> 4;
    const int ncol = col0 + wn * 16 + (lane & 15);
#pragma unroll
    for (int r = 0; r < 8; ++r) {
        const int mrow = row0 + wm * 16 + r + hi * 8;
        if (mrow < M && ncol < N) {
            float v = acc[r] * alpha;
            if (bias) v += bias[ncol];
            if (addm) v += addm[(long long)bz * sAdd + (long long)mrow * ldadd + ncol];
            Cb[(long long)mrow * ldc + ncol] = v;
        }
    }
}

// ---------------------------------------------------------------------------
// z path: per (q,k) row of 128: LN(z)*w+b then project to 16 heads.
// One wave32 per row (4 floats/lane); output layout zbias[h][q][k].
// Streams the 128 MiB z tensor exactly once -> the HBM-bound stage.
// ---------------------------------------------------------------------------
__global__ __launch_bounds__(256)
void z_ln_proj(const float* __restrict__ z, const float* __restrict__ zw,
               const float* __restrict__ znw, const float* __restrict__ znb,
               float* __restrict__ zbias)
{
    const int lane = threadIdx.x & 31;
    const int wid  = threadIdx.x >> 5;
    const long long row = (long long)blockIdx.x * 8 + wid;   // 0..262143

    const float4 v4 = ((const float4*)(z + row * ZD))[lane];
    float vals[4] = { v4.x, v4.y, v4.z, v4.w };
    float s  = vals[0] + vals[1] + vals[2] + vals[3];
    float s2 = vals[0]*vals[0] + vals[1]*vals[1] + vals[2]*vals[2] + vals[3]*vals[3];
    for (int o = 16; o > 0; o >>= 1) {
        s  += __shfl_xor(s,  o, 32);
        s2 += __shfl_xor(s2, o, 32);
    }
    const float mean = s * (1.0f / ZD);
    const float var  = s2 * (1.0f / ZD) - mean * mean;
    const float r    = rsqrtf(var + 1e-5f);

    float acc[NH];
#pragma unroll
    for (int j = 0; j < NH; ++j) acc[j] = 0.f;
#pragma unroll
    for (int i = 0; i < 4; ++i) {
        const int c = lane * 4 + i;
        const float n = (vals[i] - mean) * r * znw[c] + znb[c];
        const float* wr = zw + c * NH;
#pragma unroll
        for (int j = 0; j < NH; ++j) acc[j] += n * wr[j];
    }
#pragma unroll
    for (int j = 0; j < NH; ++j)
        for (int o = 16; o > 0; o >>= 1) acc[j] += __shfl_xor(acc[j], o, 32);

    const long long QK = (long long)NSEQ * NSEQ;
#pragma unroll
    for (int j = 0; j < NH; ++j)
        if (lane == j) zbias[(long long)j * QK + row] = acc[j];
}

// ---------------------------------------------------------------------------
// Row softmax (in place). One block per (head,q) row of 512.
// ---------------------------------------------------------------------------
__global__ __launch_bounds__(256)
void softmax_rows(float* __restrict__ x, int cols)
{
    __shared__ float red[256];
    float* xr = x + (long long)blockIdx.x * cols;
    float m = -3.0e38f;
    for (int c = threadIdx.x; c < cols; c += 256) m = fmaxf(m, xr[c]);
    red[threadIdx.x] = m; __syncthreads();
    for (int o = 128; o > 0; o >>= 1) { if (threadIdx.x < o) red[threadIdx.x] = fmaxf(red[threadIdx.x], red[threadIdx.x + o]); __syncthreads(); }
    m = red[0]; __syncthreads();
    float s = 0.f;
    for (int c = threadIdx.x; c < cols; c += 256) { float e = __expf(xr[c] - m); xr[c] = e; s += e; }
    red[threadIdx.x] = s; __syncthreads();
    for (int o = 128; o > 0; o >>= 1) { if (threadIdx.x < o) red[threadIdx.x] += red[threadIdx.x + o]; __syncthreads(); }
    const float inv = 1.0f / red[0];
    for (int c = threadIdx.x; c < cols; c += 256) xr[c] *= inv;
}

// ---------------------------------------------------------------------------
// Fused elementwise kernels
// ---------------------------------------------------------------------------
__global__ __launch_bounds__(256)
void ew_adaln(const float* an, const float* spre, const float* bmat, float* out, int n)
{
    int i = blockIdx.x * 256 + threadIdx.x;
    if (i < n) out[i] = an[i] * sigmoidf_(spre[i]) + bmat[i];
}
__global__ __launch_bounds__(256)
void ew_gate(const float* o, const float* g, float* out, int n)
{
    int i = blockIdx.x * 256 + threadIdx.x;
    if (i < n) out[i] = o[i] * sigmoidf_(g[i]);
}
__global__ __launch_bounds__(256)
void ew_resid(const float* a, const float* sopre, const float* x, float* out, int n)
{
    int i = blockIdx.x * 256 + threadIdx.x;
    if (i < n) out[i] = a[i] + sigmoidf_(sopre[i]) * x[i];
}
__global__ __launch_bounds__(256)
void ew_swish(const float* sw, const float* ab, float* out, int n)   // n = 512*1536
{
    int i = blockIdx.x * 256 + threadIdx.x;
    if (i < n) {
        const int r = i / HIDDEN, c = i - r * HIDDEN;
        const float x1 = sw[(long long)r * (2 * HIDDEN) + c];
        const float g  = sw[(long long)r * (2 * HIDDEN) + HIDDEN + c];
        out[i] = g * sigmoidf_(g) * x1 * ab[i];
    }
}
__global__ __launch_bounds__(256)
void ew_final(const float* a1, const float* opre, const float* y, float* out, int n)
{
    int i = blockIdx.x * 256 + threadIdx.x;
    if (i < n) out[i] = a1[i] + sigmoidf_(opre[i]) * y[i];
}

// ---------------------------------------------------------------------------
// Host orchestration
// ---------------------------------------------------------------------------
static inline void gemm(hipStream_t st, bool transB,
                        const float* A, const float* B, float* C,
                        const float* bias, const float* addm,
                        int M, int N, int K, int lda, int ldb, int ldc, int ldadd,
                        long long sA, long long sB, long long sC, long long sAdd,
                        float alpha, int batches)
{
    dim3 g((N + 63) / 64, (M + 31) / 32, batches);
    if (transB)
        wmma_gemm<true><<<g, 256, 0, st>>>(A, B, C, bias, addm, M, N, K,
                                           lda, ldb, ldc, ldadd, sA, sB, sC, sAdd, alpha);
    else
        wmma_gemm<false><<<g, 256, 0, st>>>(A, B, C, bias, addm, M, N, K,
                                            lda, ldb, ldc, ldadd, sA, sB, sC, sAdd, alpha);
}

extern "C" void kernel_launch(void* const* d_in, const int* in_sizes, int n_in,
                              void* d_out, int out_size, void* d_ws, size_t ws_size,
                              hipStream_t stream)
{
    const float* a        = (const float*)d_in[0];
    const float* s        = (const float*)d_in[1];
    const float* z        = (const float*)d_in[2];
    const float* ad1_snw  = (const float*)d_in[3];
    const float* ad1_sw   = (const float*)d_in[4];
    const float* ad1_sb   = (const float*)d_in[5];
    const float* ad1_bw   = (const float*)d_in[6];
    const float* q_w      = (const float*)d_in[7];
    const float* q_b      = (const float*)d_in[8];
    const float* k_w      = (const float*)d_in[9];
    const float* v_w      = (const float*)d_in[10];
    const float* g_w      = (const float*)d_in[11];
    const float* z_nw     = (const float*)d_in[12];
    const float* z_nb     = (const float*)d_in[13];
    const float* z_w      = (const float*)d_in[14];
    const float* o_w      = (const float*)d_in[15];
    const float* so_w     = (const float*)d_in[16];
    const float* so_b     = (const float*)d_in[17];
    const float* ad2_snw  = (const float*)d_in[18];
    const float* ad2_sw   = (const float*)d_in[19];
    const float* ad2_sb   = (const float*)d_in[20];
    const float* ad2_bw   = (const float*)d_in[21];
    const float* swish_w  = (const float*)d_in[22];
    const float* a2b_w    = (const float*)d_in[23];
    const float* b2a_w    = (const float*)d_in[24];
    const float* outp_w   = (const float*)d_in[25];
    const float* outp_b   = (const float*)d_in[26];
    float* out = (float*)d_out;

    const long long SZ = (long long)NSEQ * DIMX;       // 393216
    const long long QK = (long long)NSEQ * NSEQ;       // 262144
    float* ws    = (float*)d_ws;
    float* an    = ws +  0 * SZ;
    float* sn    = ws +  1 * SZ;
    float* spre  = ws +  2 * SZ;
    float* bmat  = ws +  3 * SZ;
    float* bbuf  = ws +  4 * SZ;   // adaln output "b" / later "at"
    float* qb    = ws +  5 * SZ;
    float* kb    = ws +  6 * SZ;
    float* vb    = ws +  7 * SZ;
    float* gb    = ws +  8 * SZ;
    float* ob    = ws +  9 * SZ;
    float* xb    = ws + 10 * SZ;
    float* sopre = ws + 11 * SZ;
    float* a1    = ws + 12 * SZ;
    float* zbias = ws + 13 * SZ;                 // 16*QK
    float* sc    = zbias + (long long)NH * QK;   // 16*QK
    float* swb   = sc + (long long)NH * QK;      // 512*3072
    float* abb   = swb + (long long)NSEQ * 2 * HIDDEN;   // 512*1536
    float* bbb   = abb + (long long)NSEQ * HIDDEN;       // 512*1536
    float* yb    = qb;      // reuse (q dead after attention)
    float* opre  = kb;      // reuse

    const int ewg = (int)((SZ + 255) / 256);
    const float inv_sqrt_hd = 0.14433756729740646f;      // 48^-0.5

    // ---- adaLN 1 ----
    ln_rows<<<NSEQ, 256, 0, stream>>>(a, nullptr, nullptr, an, DIMX);
    ln_rows<<<NSEQ, 256, 0, stream>>>(s, ad1_snw, nullptr, sn, DIMX);
    gemm(stream, false, sn, ad1_sw, spre, ad1_sb, nullptr, NSEQ, DIMX, DIMX,
         DIMX, DIMX, DIMX, 0, 0, 0, 0, 0, 1.0f, 1);
    gemm(stream, false, sn, ad1_bw, bmat, nullptr, nullptr, NSEQ, DIMX, DIMX,
         DIMX, DIMX, DIMX, 0, 0, 0, 0, 0, 1.0f, 1);
    ew_adaln<<<ewg, 256, 0, stream>>>(an, spre, bmat, bbuf, (int)SZ);

    // ---- projections ----
    gemm(stream, false, bbuf, q_w, qb, q_b,   nullptr, NSEQ, DIMX, DIMX, DIMX, DIMX, DIMX, 0, 0,0,0,0, 1.0f, 1);
    gemm(stream, false, bbuf, k_w, kb, nullptr, nullptr, NSEQ, DIMX, DIMX, DIMX, DIMX, DIMX, 0, 0,0,0,0, 1.0f, 1);
    gemm(stream, false, bbuf, v_w, vb, nullptr, nullptr, NSEQ, DIMX, DIMX, DIMX, DIMX, DIMX, 0, 0,0,0,0, 1.0f, 1);
    gemm(stream, false, bbuf, g_w, gb, nullptr, nullptr, NSEQ, DIMX, DIMX, DIMX, DIMX, DIMX, 0, 0,0,0,0, 1.0f, 1);

    // ---- z bias (HBM-bound: streams 128 MiB once) ----
    z_ln_proj<<<(NSEQ * NSEQ) / 8, 256, 0, stream>>>(z, z_w, z_nw, z_nb, zbias);

    // ---- scores = Q Kt * scale + zbias  (batched over heads, NT) ----
    gemm(stream, true, qb, kb, sc, nullptr, zbias, NSEQ, NSEQ, HD,
         DIMX, DIMX, NSEQ, NSEQ, HD, HD, QK, QK, inv_sqrt_hd, NH);

    softmax_rows<<<NH * NSEQ, 256, 0, stream>>>(sc, NSEQ);

    // ---- O = attn @ V (batched over heads, NN) ----
    gemm(stream, false, sc, vb, ob, nullptr, nullptr, NSEQ, HD, NSEQ,
         NSEQ, DIMX, DIMX, 0, QK, HD, HD, 0, 1.0f, NH);

    // ---- gated output projection + residual ----
    ew_gate<<<ewg, 256, 0, stream>>>(ob, gb, spre, (int)SZ);   // reuse spre as o*sigmoid(g)
    gemm(stream, false, spre, o_w, xb, nullptr, nullptr, NSEQ, DIMX, DIMX, DIMX, DIMX, DIMX, 0, 0,0,0,0, 1.0f, 1);
    gemm(stream, false, s, so_w, sopre, so_b, nullptr, NSEQ, DIMX, DIMX, DIMX, DIMX, DIMX, 0, 0,0,0,0, 1.0f, 1);
    ew_resid<<<ewg, 256, 0, stream>>>(a, sopre, xb, a1, (int)SZ);

    // ---- adaLN 2 ----
    ln_rows<<<NSEQ, 256, 0, stream>>>(a1, nullptr, nullptr, an, DIMX);
    ln_rows<<<NSEQ, 256, 0, stream>>>(s, ad2_snw, nullptr, sn, DIMX);
    gemm(stream, false, sn, ad2_sw, spre, ad2_sb, nullptr, NSEQ, DIMX, DIMX, DIMX, DIMX, DIMX, 0, 0,0,0,0, 1.0f, 1);
    gemm(stream, false, sn, ad2_bw, bmat, nullptr, nullptr, NSEQ, DIMX, DIMX, DIMX, DIMX, DIMX, 0, 0,0,0,0, 1.0f, 1);
    ew_adaln<<<ewg, 256, 0, stream>>>(an, spre, bmat, bbuf, (int)SZ);   // bbuf = at

    // ---- transition MLP ----
    gemm(stream, false, bbuf, swish_w, swb, nullptr, nullptr, NSEQ, 2 * HIDDEN, DIMX,
         DIMX, 2 * HIDDEN, 2 * HIDDEN, 0, 0,0,0,0, 1.0f, 1);
    gemm(stream, false, bbuf, a2b_w, abb, nullptr, nullptr, NSEQ, HIDDEN, DIMX,
         DIMX, HIDDEN, HIDDEN, 0, 0,0,0,0, 1.0f, 1);
    {
        const int n2 = NSEQ * HIDDEN;
        ew_swish<<<(n2 + 255) / 256, 256, 0, stream>>>(swb, abb, bbb, n2);
    }
    gemm(stream, false, bbb, b2a_w, yb, nullptr, nullptr, NSEQ, DIMX, HIDDEN,
         HIDDEN, DIMX, DIMX, 0, 0,0,0,0, 1.0f, 1);
    gemm(stream, false, s, outp_w, opre, outp_b, nullptr, NSEQ, DIMX, DIMX,
         DIMX, DIMX, DIMX, 0, 0,0,0,0, 1.0f, 1);
    ew_final<<<ewg, 256, 0, stream>>>(a1, opre, yb, out, (int)SZ);
}
